// FusedTriMulOutInWrapper_83708912599797
// MI455X (gfx1250) — matrix operand order repaired and checked by reading, not compiled, not run on hardware
//
#include <hip/hip_runtime.h>
#include <hip/hip_bf16.h>
#include <cstdint>

#define NDIM 384
#define NT   (NDIM * NDIM)   // 147456 positions
#define CC   128             // channels (CZ == CH == 128)

typedef __bf16 bf16_t;
typedef __attribute__((ext_vector_type(16))) __bf16 v16bf;
typedef __attribute__((ext_vector_type(8)))  __bf16 v8bf;
typedef __attribute__((ext_vector_type(8)))  float  v8f;
typedef unsigned int u32x4 __attribute__((ext_vector_type(4)));
typedef int          i32x8 __attribute__((ext_vector_type(8)));
typedef int          i32x4 __attribute__((ext_vector_type(4)));

#if __has_builtin(__builtin_amdgcn_tensor_load_to_lds) && __has_builtin(__builtin_amdgcn_s_wait_tensorcnt)
#define USE_TDM 1
#else
#define USE_TDM 0
#endif

__device__ __forceinline__ float sigmoid_f(float x) {
    return 1.0f / (1.0f + __expf(-x));
}

__device__ __forceinline__ v16bf cat16(v8bf lo, v8bf hi) {
    return __builtin_shufflevector(lo, hi, 0, 1, 2, 3, 4, 5, 6, 7,
                                           8, 9, 10, 11, 12, 13, 14, 15);
}

// ---------------------------------------------------------------------------
// WMMA fragment loaders (wave32, 16x16x32 bf16; layouts per ISA 7.12.2)
// A 16x32 (MxK): lane l -> M = l&15, K-half = l>>4; halfs [0..7] = Kbase+0..7,
//                halfs [8..15] = Kbase+16..23 (two contiguous 8-elem runs).
// B 32x16 (KxN): lane l -> N col = l&15, K-half = l>>4; same K order as A.
// ---------------------------------------------------------------------------
__device__ __forceinline__ v16bf afrag_any(const bf16_t* tile, int ld, int kb, int lane) {
    const int m   = lane & 15;
    const int khi = (lane >> 4) & 1;
    const bf16_t* r = tile + (long)m * ld + kb * 32 + khi * 8;
    return cat16(*(const v8bf*)(r), *(const v8bf*)(r + 16));
}

// B^T-style fragment: column n of B^T == row (n0+n) of a row-major matrix.
__device__ __forceinline__ v16bf bfrag_rows(const bf16_t* base, int ld, int kb, int n0, int lane) {
    const int n  = lane & 15;
    const int kg = (lane >> 4) & 1;
    const bf16_t* r = base + (long)(n0 + n) * ld + kb * 32 + kg * 8;
    return cat16(*(const v8bf*)(r), *(const v8bf*)(r + 16));
}

// ---------------------------------------------------------------------------
// Tensor Data Mover: async 2-D tile (global -> LDS), D# per ISA ch.8.
// tile_x contiguous elems (2B each) x tile_y rows, row stride = stride elems.
// ---------------------------------------------------------------------------
__device__ __forceinline__ void tdm_load_2d(const bf16_t* gsrc, unsigned lds_off,
                                            int tile_x, int tile_y, int stride) {
#if USE_TDM
    unsigned long long ga = (unsigned long long)(size_t)gsrc;
    u32x4 g0;
    g0[0] = 1u;                                   // count=1, user descriptor
    g0[1] = lds_off;                              // lds_addr (bytes)
    g0[2] = (unsigned)(ga & 0xffffffffu);         // global_addr[31:0]
    g0[3] = (unsigned)((ga >> 32) & 0x01ffffffu)  // global_addr[56:32]
          | (2u << 30);                           // type = 2 ("image")
    i32x8 g1;
    g1[0] = (1 << 16);                            // data_size = 1 -> 2 bytes
    g1[1] = (int)(((unsigned)stride & 0xffffu) << 16);        // tensor_dim0 lo16
    g1[2] = (int)((((unsigned)stride >> 16) & 0xffffu)
          | (((unsigned)NDIM & 0xffffu) << 16));              // dim0 hi | dim1 lo
    g1[3] = (int)((((unsigned)NDIM >> 16) & 0xffffu)
          | (((unsigned)tile_x & 0xffffu) << 16));            // dim1 hi | tile_dim0
    g1[4] = tile_y & 0xffff;                      // tile_dim1 (tile_dim2 = 0)
    g1[5] = stride;                               // tensor_dim0_stride[31:0]
    g1[6] = 0;                                    // stride hi | dim1_stride lo
    g1[7] = 0;
    i32x4 gz4 = {0, 0, 0, 0};
    i32x8 gz8 = {0, 0, 0, 0, 0, 0, 0, 0};
    __builtin_amdgcn_tensor_load_to_lds(g0, g1, gz4, gz4, gz8, 0);
#endif
}

// ---------------------------------------------------------------------------
// Weight convert+transpose f32 [K][N] -> bf16 [N][K], order: ag,ap,bg,bp,g,z
// (transposed so WMMA B-fragments become contiguous row loads)
// ---------------------------------------------------------------------------
__global__ __launch_bounds__(256) void convert_weights_kernel(
    const float* __restrict__ s0, const float* __restrict__ s1,
    const float* __restrict__ s2, const float* __restrict__ s3,
    const float* __restrict__ s4, const float* __restrict__ s5,
    bf16_t* __restrict__ dst)
{
    const int i   = blockIdx.x * 256 + threadIdx.x;
    const int per = 2 * CC * CC;
    if (i >= 6 * per) return;
    const int r = i / per, t = i % per;
    const int h = t / (CC * CC), rem = t % (CC * CC);
    const int n = rem / CC, k = rem % CC;
    const float* s = (r == 0) ? s0 : (r == 1) ? s1 : (r == 2) ? s2
                   : (r == 3) ? s3 : (r == 4) ? s4 : s5;
    dst[i] = (bf16_t)s[h * CC * CC + k * CC + n];
}

// ---------------------------------------------------------------------------
// Kernel 1: zn = LN(z); a = sig(zn@Wag+bag)*(zn@Wap+bap);
//           b = sig(zn@Wbg+bbg)*(zn@Wbp+bbp); g = sig(zn@Wg+bg)
// Block = 16 positions x 128 channels, 8 waves (one 16-col tile each).
// a,b stored channel-major [c][N][N], spatially transposed when `transposed`.
// ---------------------------------------------------------------------------
__global__ __launch_bounds__(256) void tri_proj_kernel(
    const float* __restrict__ zin,
    const float* __restrict__ lnw, const float* __restrict__ lnb,
    const bf16_t* __restrict__ Wag, const bf16_t* __restrict__ Wap,
    const bf16_t* __restrict__ Wbg, const bf16_t* __restrict__ Wbp,
    const bf16_t* __restrict__ Wg,
    const float* __restrict__ bag, const float* __restrict__ bap,
    const float* __restrict__ bbg, const float* __restrict__ bbp,
    const float* __restrict__ bgg,
    bf16_t* __restrict__ Aout, bf16_t* __restrict__ Bout,
    float* __restrict__ Gout, int transposed)
{
    __shared__ __align__(16) bf16_t zn[16 * CC];
    __shared__ float stats[32];
    const int tid  = threadIdx.x;
    const int pos0 = blockIdx.x * 16;
    const float* ztile = zin + (long)pos0 * CC;
    __builtin_prefetch(ztile + 16 * CC, 0, 0);     // next block's tile

    if (tid < 16) {
        const float4* row = (const float4*)(ztile + tid * CC);
        float s = 0.f, s2 = 0.f;
#pragma unroll
        for (int c = 0; c < CC / 4; ++c) {
            float4 v = row[c];
            s  += v.x + v.y + v.z + v.w;
            s2 += v.x * v.x + v.y * v.y + v.z * v.z + v.w * v.w;
        }
        float m   = s * (1.0f / CC);
        float var = s2 * (1.0f / CC) - m * m;
        stats[tid]      = m;
        stats[16 + tid] = rsqrtf(var + 1e-5f);
    }
    __syncthreads();
#pragma unroll
    for (int e = 0; e < 8; ++e) {
        int i = e * 256 + tid;
        int r = i >> 7, c = i & 127;
        zn[i] = (bf16_t)((ztile[i] - stats[r]) * stats[16 + r] * lnw[c] + lnb[c]);
    }
    __syncthreads();

    const int lane = tid & 31, wv = tid >> 5;
    const int n0   = wv * 16;

    v16bf afr[4];
#pragma unroll
    for (int kb = 0; kb < 4; ++kb) afr[kb] = afrag_any(zn, CC, kb, lane);

    const int ccol = n0 + (lane & 15);
    const int i_sp = pos0 / NDIM, j0 = pos0 % NDIM;

    // ---- a ----
    {
        v8f acc1 = {}, acc2 = {};
#pragma unroll
        for (int kb = 0; kb < 4; ++kb) {
            acc1 = __builtin_amdgcn_wmma_f32_16x16x32_bf16(
                false, afr[kb], false, bfrag_rows(Wag, CC, kb, n0, lane), (short)0, acc1, false, false);
            acc2 = __builtin_amdgcn_wmma_f32_16x16x32_bf16(
                false, afr[kb], false, bfrag_rows(Wap, CC, kb, n0, lane), (short)0, acc2, false, false);
        }
        const float c1 = bag[ccol], c2 = bap[ccol];
#pragma unroll
        for (int v = 0; v < 8; ++v) {
            const int M  = v + 8 * (lane >> 4);
            const int jj = j0 + M;
            const float a = sigmoid_f(acc1[v] + c1) * (acc2[v] + c2);
            const long  o = transposed ? ((long)ccol * NT + (long)jj * NDIM + i_sp)
                                       : ((long)ccol * NT + (long)i_sp * NDIM + jj);
            Aout[o] = (bf16_t)a;
        }
    }
    // ---- b ----
    {
        v8f acc1 = {}, acc2 = {};
#pragma unroll
        for (int kb = 0; kb < 4; ++kb) {
            acc1 = __builtin_amdgcn_wmma_f32_16x16x32_bf16(
                false, afr[kb], false, bfrag_rows(Wbg, CC, kb, n0, lane), (short)0, acc1, false, false);
            acc2 = __builtin_amdgcn_wmma_f32_16x16x32_bf16(
                false, afr[kb], false, bfrag_rows(Wbp, CC, kb, n0, lane), (short)0, acc2, false, false);
        }
        const float c1 = bbg[ccol], c2 = bbp[ccol];
#pragma unroll
        for (int v = 0; v < 8; ++v) {
            const int M  = v + 8 * (lane >> 4);
            const int jj = j0 + M;
            const float b = sigmoid_f(acc1[v] + c1) * (acc2[v] + c2);
            const long  o = transposed ? ((long)ccol * NT + (long)jj * NDIM + i_sp)
                                       : ((long)ccol * NT + (long)i_sp * NDIM + jj);
            Bout[o] = (bf16_t)b;
        }
    }
    // ---- g (position-major) ----
    {
        v8f acc = {};
#pragma unroll
        for (int kb = 0; kb < 4; ++kb) {
            acc = __builtin_amdgcn_wmma_f32_16x16x32_bf16(
                false, afr[kb], false, bfrag_rows(Wg, CC, kb, n0, lane), (short)0, acc, false, false);
        }
        const float cb = bgg[ccol];
#pragma unroll
        for (int v = 0; v < 8; ++v) {
            const int M = v + 8 * (lane >> 4);
            Gout[(long)(pos0 + M) * CC + ccol] = sigmoid_f(acc[v] + cb);
        }
    }
}

// ---------------------------------------------------------------------------
// Kernel 2: per-channel triangle einsum  P_c = A_c * B_c^T  (384x384 each).
// Serves both directions (tri_proj stores transposed for incoming).
// Block = one 64x64 tile of one channel; per K-step the 64x32 A and B panels
// are DMA'd into LDS by the Tensor Data Mover (TENSORcnt), then 8 waves run
// 2 WMMAs each off LDS fragments.
// ---------------------------------------------------------------------------
__global__ __launch_bounds__(256) void tri_einsum_kernel(
    const bf16_t* __restrict__ Am, const bf16_t* __restrict__ Bm,
    float* __restrict__ Pout)
{
    __shared__ __align__(16) bf16_t As[64 * 32];
    __shared__ __align__(16) bf16_t Bs[64 * 32];

    const int c  = blockIdx.y;
    const int tx = blockIdx.x % (NDIM / 64);
    const int ty = blockIdx.x / (NDIM / 64);
    const int m0 = ty * 64, nb = tx * 64;

    const bf16_t* Ac = Am + (long)c * NT;
    const bf16_t* Bc = Bm + (long)c * NT;

    const int tid = threadIdx.x, lane = tid & 31, wv = tid >> 5;
    const int wm = wv & 3, wn = wv >> 2;

#if USE_TDM
    const unsigned as_off = (unsigned)(size_t)(void*)As;
    const unsigned bs_off = (unsigned)(size_t)(void*)Bs;
#endif

    v8f acc0 = {}, acc1 = {};
    for (int kb = 0; kb < 12; ++kb) {
        __syncthreads();   // previous iteration's LDS reads done
#if USE_TDM
        if (wv == 0) {
            tdm_load_2d(Ac + (long)m0 * NDIM + kb * 32, as_off, 32, 64, NDIM);
            tdm_load_2d(Bc + (long)nb * NDIM + kb * 32, bs_off, 32, 64, NDIM);
            __builtin_amdgcn_s_wait_tensorcnt(0);
        }
#else
        {
            const int row = tid >> 2, seg = tid & 3;
            *(v8bf*)(As + row * 32 + seg * 8) =
                *(const v8bf*)(Ac + (long)(m0 + row) * NDIM + kb * 32 + seg * 8);
            *(v8bf*)(Bs + row * 32 + seg * 8) =
                *(const v8bf*)(Bc + (long)(nb + row) * NDIM + kb * 32 + seg * 8);
        }
#endif
        __syncthreads();
        v16bf af  = afrag_any(As + wm * 16 * 32, 32, 0, lane);
        v16bf bf0 = bfrag_rows(Bs, 32, 0, wn * 32, lane);
        v16bf bf1 = bfrag_rows(Bs, 32, 0, wn * 32 + 16, lane);
        acc0 = __builtin_amdgcn_wmma_f32_16x16x32_bf16(false, af, false, bf0, (short)0, acc0, false, false);
        acc1 = __builtin_amdgcn_wmma_f32_16x16x32_bf16(false, af, false, bf1, (short)0, acc1, false, false);
    }
#pragma unroll
    for (int v = 0; v < 8; ++v) {
        const int i  = m0 + wm * 16 + v + 8 * (lane >> 4);
        const int k0 = nb + wn * 32 + (lane & 15);
        Pout[((long)i * NDIM + k0) * CC + c]      = acc0[v];
        Pout[((long)i * NDIM + k0 + 16) * CC + c] = acc1[v];
    }
}

// ---------------------------------------------------------------------------
// Kernel 3: out = z + (LayerNorm(P) @ Wz + bz) * g
// ---------------------------------------------------------------------------
__global__ __launch_bounds__(256) void tri_out_kernel(
    const float* __restrict__ Pin,
    const float* __restrict__ lnw, const float* __restrict__ lnb,
    const bf16_t* __restrict__ Wz, const float* __restrict__ bz,
    const float* __restrict__ G, const float* __restrict__ Zres,
    float* __restrict__ Zout)
{
    __shared__ __align__(16) bf16_t pn[16 * CC];
    __shared__ float stats[32];
    const int tid  = threadIdx.x;
    const int pos0 = blockIdx.x * 16;
    const float* ptile = Pin + (long)pos0 * CC;

    if (tid < 16) {
        const float4* row = (const float4*)(ptile + tid * CC);
        float s = 0.f, s2 = 0.f;
#pragma unroll
        for (int c = 0; c < CC / 4; ++c) {
            float4 v = row[c];
            s  += v.x + v.y + v.z + v.w;
            s2 += v.x * v.x + v.y * v.y + v.z * v.z + v.w * v.w;
        }
        float m   = s * (1.0f / CC);
        float var = s2 * (1.0f / CC) - m * m;
        stats[tid]      = m;
        stats[16 + tid] = rsqrtf(var + 1e-5f);
    }
    __syncthreads();
#pragma unroll
    for (int e = 0; e < 8; ++e) {
        int i = e * 256 + tid;
        int r = i >> 7, c = i & 127;
        pn[i] = (bf16_t)((ptile[i] - stats[r]) * stats[16 + r] * lnw[c] + lnb[c]);
    }
    __syncthreads();

    const int lane = tid & 31, wv = tid >> 5;
    const int n0 = wv * 16;

    v8f acc = {};
#pragma unroll
    for (int kb = 0; kb < 4; ++kb) {
        acc = __builtin_amdgcn_wmma_f32_16x16x32_bf16(
            false, afrag_any(pn, CC, kb, lane), false,
            bfrag_rows(Wz, CC, kb, n0, lane), (short)0, acc, false, false);
    }
    const int   ccol = n0 + (lane & 15);
    const float bzc  = bz[ccol];
#pragma unroll
    for (int v = 0; v < 8; ++v) {
        const int  pos = pos0 + v + 8 * (lane >> 4);
        const long o   = (long)pos * CC + ccol;
        Zout[o] = Zres[o] + (acc[v] + bzc) * G[o];
    }
}

// ---------------------------------------------------------------------------
// Host launcher
// ---------------------------------------------------------------------------
extern "C" void kernel_launch(void* const* d_in, const int* in_sizes, int n_in,
                              void* d_out, int out_size, void* d_ws, size_t ws_size,
                              hipStream_t stream)
{
    const float* z0       = (const float*)d_in[0];
    const float* ln_in_w  = (const float*)d_in[1];
    const float* ln_in_b  = (const float*)d_in[2];
    const float* w_ag     = (const float*)d_in[3];
    const float* b_ag     = (const float*)d_in[4];
    const float* w_ap     = (const float*)d_in[5];
    const float* b_ap     = (const float*)d_in[6];
    const float* w_bg     = (const float*)d_in[7];
    const float* b_bg     = (const float*)d_in[8];
    const float* w_bp     = (const float*)d_in[9];
    const float* b_bp     = (const float*)d_in[10];
    const float* ln_out_w = (const float*)d_in[11];
    const float* ln_out_b = (const float*)d_in[12];
    const float* w_z      = (const float*)d_in[13];
    const float* b_z      = (const float*)d_in[14];
    const float* w_g      = (const float*)d_in[15];
    const float* b_g      = (const float*)d_in[16];

    char* ws = (char*)d_ws;
    size_t o = 0;
    const size_t WMAT = (size_t)2 * CC * CC;                  // 32768 elems / tensor
    bf16_t* Wbf = (bf16_t*)(ws + o); o += 6 * WMAT * sizeof(bf16_t);
    o = (o + 255) & ~(size_t)255;
    float*  Z1   = (float*)(ws + o);  o += (size_t)NT * CC * sizeof(float);
    bf16_t* Abuf = (bf16_t*)(ws + o); o += (size_t)CC * NT * sizeof(bf16_t);
    bf16_t* Bbuf = (bf16_t*)(ws + o); o += (size_t)CC * NT * sizeof(bf16_t);
    float*  Gbuf = (float*)(ws + o);  o += (size_t)NT * CC * sizeof(float);
    float*  Pbuf = (float*)(ws + o);  o += (size_t)NT * CC * sizeof(float);

    bf16_t* Wag = Wbf + 0 * WMAT;   // all stored transposed [N][K] in bf16
    bf16_t* Wap = Wbf + 1 * WMAT;
    bf16_t* Wbg = Wbf + 2 * WMAT;
    bf16_t* Wbp = Wbf + 3 * WMAT;
    bf16_t* Wg  = Wbf + 4 * WMAT;
    bf16_t* Wz  = Wbf + 5 * WMAT;

    const int nconv = (int)(6 * WMAT);
    convert_weights_kernel<<<(nconv + 255) / 256, 256, 0, stream>>>(
        w_ag, w_ap, w_bg, w_bp, w_g, w_z, Wbf);

    for (int idx = 0; idx < 2; ++idx) {
        const float* zsrc = (idx == 0) ? z0 : Z1;
        float*       zdst = (idx == 0) ? Z1 : (float*)d_out;
        const size_t wo   = (size_t)idx * CC * CC;

        tri_proj_kernel<<<NT / 16, 256, 0, stream>>>(
            zsrc, ln_in_w + idx * CC, ln_in_b + idx * CC,
            Wag + wo, Wap + wo, Wbg + wo, Wbp + wo, Wg + wo,
            b_ag + idx * CC, b_ap + idx * CC, b_bg + idx * CC,
            b_bp + idx * CC, b_g + idx * CC,
            Abuf, Bbuf, Gbuf, /*transposed=*/idx);

        tri_einsum_kernel<<<dim3((NDIM / 64) * (NDIM / 64), CC), 256, 0, stream>>>(
            Abuf, Bbuf, Pbuf);

        tri_out_kernel<<<NT / 16, 256, 0, stream>>>(
            Pbuf, ln_out_w + idx * CC, ln_out_b + idx * CC,
            Wz + wo, b_z + idx * CC, Gbuf, zsrc, zdst);
    }
}
